// LagAwareEnergyModel_20083267076445
// MI455X (gfx1250) — matrix operand order repaired and checked by reading, not compiled, not run on hardware
//
#include <hip/hip_runtime.h>
#include <hip/hip_bf16.h>

// ---------------------------------------------------------------------------
// Problem constants (match reference)
// ---------------------------------------------------------------------------
static constexpr int Qn  = 512;
static constexpr int Cn  = 512;
static constexpr int Tn  = 128;
static constexpr int DIN = 512;
static constexpr int DH  = 256;
static constexpr int NLAG = 9;          // max_lag = 4 -> deltas -4..4
static constexpr int NROWS = Qn * Tn;   // 65536 rows for both MLPs

typedef __attribute__((ext_vector_type(16))) __bf16 v16bf;
typedef __attribute__((ext_vector_type(8)))  __bf16 v8bf;
typedef __attribute__((ext_vector_type(8)))  float  v8f;

struct BfPair { v8bf lo; v8bf hi; };   // 32 bytes -> bit_cast to v16bf

__device__ __forceinline__ __bf16 f2bf(float f) {
    unsigned u = __builtin_bit_cast(unsigned, f);
    unsigned r = (u + 0x7FFFu + ((u >> 16) & 1u)) >> 16;   // round-nearest-even
    unsigned short s = (unsigned short)r;
    return __builtin_bit_cast(__bf16, s);
}

// ---------------------------------------------------------------------------
// Kernel 1: fold projection W into video layer-2:  Wc = v_w2 @ W, bc = v_b2 @ W
// ---------------------------------------------------------------------------
__global__ void fuse_w_kernel(const float* __restrict__ v_w2,
                              const float* __restrict__ v_b2,
                              const float* __restrict__ W,
                              float* __restrict__ Wc,
                              float* __restrict__ bc) {
    int i = blockIdx.x;    // 0..255  (row of v_w2)
    int j = threadIdx.x;   // 0..255  (col of W)
    float s = 0.f;
    for (int k = 0; k < DH; ++k) s += v_w2[i * DH + k] * W[k * DH + j];
    Wc[i * DH + j] = s;
    if (i == 0) {
        float sb = 0.f;
        for (int k = 0; k < DH; ++k) sb += v_b2[k] * W[k * DH + j];
        bc[j] = sb;
    }
}

// ---------------------------------------------------------------------------
// Kernel 2: tiled bf16-WMMA GEMM  Out[bf16, M x 256] = act(X[M x K] @ Wm[K x 256] + bias)
//   Block: 256 threads = 8 waves.  Tile: 64(M) x 64(N), K-step 32.
//   Wave w: m-subtile = w%4, two n-subtiles = 2*(w/4), 2*(w/4)+1.
// ---------------------------------------------------------------------------
template <bool XBF16, bool RELU>
__global__ void __launch_bounds__(256)
gemm_bias_act(const void*  __restrict__ Xv,
              const float* __restrict__ Wm,
              const float* __restrict__ bias,
              __bf16*      __restrict__ Out,
              int M, int K) {
    constexpr int BM = 64, BN = 64, BK = 32;
    __shared__ alignas(32) __bf16 As[BM * BK];   // [m][k] row-major
    __shared__ alignas(32) __bf16 Bs[BN * BK];   // [n][k] (K-transposed)

    const int tid   = threadIdx.x;
    const int lane  = tid & 31;
    const int wv    = tid >> 5;
    const int m0    = blockIdx.x * BM;
    const int n0    = blockIdx.y * BN;
    const int m_sub = wv & 3;
    const int nbase = (wv >> 2) * 2;
    const int half  = lane >> 4;
    const int l16   = lane & 15;

    v8f acc0 = {};
    v8f acc1 = {};

    for (int k0 = 0; k0 < K; k0 += BK) {
        // ---- stage A tile: 64x32 elements, 8 per thread --------------------
        {
            const int row = tid >> 2;            // 0..63
            const int kk  = (tid & 3) * 8;       // 0,8,16,24
            if (XBF16) {
                const __bf16* X = (const __bf16*)Xv;
                *(v8bf*)(As + row * BK + kk) =
                    *(const v8bf*)(X + (size_t)(m0 + row) * K + k0 + kk);
            } else {
                const float* X = (const float*)Xv;
                const float4* src = (const float4*)(X + (size_t)(m0 + row) * K + k0 + kk);
                float4 f0 = src[0], f1 = src[1];
                v8bf b;
                b[0] = f2bf(f0.x); b[1] = f2bf(f0.y); b[2] = f2bf(f0.z); b[3] = f2bf(f0.w);
                b[4] = f2bf(f1.x); b[5] = f2bf(f1.y); b[6] = f2bf(f1.z); b[7] = f2bf(f1.w);
                *(v8bf*)(As + row * BK + kk) = b;
            }
        }
        // ---- stage B tile transposed: Wm rows k0..k0+31, cols n0..n0+63 ----
        #pragma unroll
        for (int it = 0; it < 2; ++it) {
            const int k  = tid >> 3;                      // 0..31
            const int nl = (tid & 7) * 4 + it * 32;       // 0..60
            float4 f = *(const float4*)(Wm + (size_t)(k0 + k) * DH + n0 + nl);
            Bs[(nl + 0) * BK + k] = f2bf(f.x);
            Bs[(nl + 1) * BK + k] = f2bf(f.y);
            Bs[(nl + 2) * BK + k] = f2bf(f.z);
            Bs[(nl + 3) * BK + k] = f2bf(f.w);
        }
        __syncthreads();

        // ---- fragments + WMMA ---------------------------------------------
        // A frag: lane holds row m=l16, K = 16*half .. +15 contiguous
        const __bf16* arow = As + (m_sub * 16 + l16) * BK + half * 16;
        BfPair ap;
        ap.lo = *(const v8bf*)(arow);
        ap.hi = *(const v8bf*)(arow + 8);
        v16bf av = __builtin_bit_cast(v16bf, ap);
        // B frags: lane holds col n=l16, K = 16*half .. +15 contiguous
        v16bf bv0 = *(const v16bf*)(Bs + ((nbase + 0) * 16 + l16) * BK + half * 16);
        v16bf bv1 = *(const v16bf*)(Bs + ((nbase + 1) * 16 + l16) * BK + half * 16);

        acc0 = __builtin_amdgcn_wmma_f32_16x16x32_bf16(false, av, false, bv0,
                                                       (short)0, acc0, false, false);
        acc1 = __builtin_amdgcn_wmma_f32_16x16x32_bf16(false, av, false, bv1,
                                                       (short)0, acc1, false, false);
        __syncthreads();
    }

    // ---- epilogue: bias (+relu), convert, store bf16 -----------------------
    #pragma unroll
    for (int s = 0; s < 2; ++s) {
        v8f acc = s ? acc1 : acc0;
        const int n = n0 + (nbase + s) * 16 + l16;
        const float bv = bias[n];
        #pragma unroll
        for (int r = 0; r < 8; ++r) {
            const int m = m0 + m_sub * 16 + r + 8 * half;
            float val = acc[r] + bv;
            if (RELU) val = fmaxf(val, 0.f);
            Out[(size_t)m * DH + n] = f2bf(val);
        }
    }
}

// ---------------------------------------------------------------------------
// Kernel 3: lag-max correlation.
//   best[q,c] = max_{Δ in -4..4} Σ_t Σ_d a[q,t,d] * pv[c,(t-Δ)&127,d]
//   Block: 128 threads = 4 waves, tile 32q x 32c (2x2 waves of 16q x 16c).
//   Per wave: 9 lag accumulators (72 VGPRs) -- fits the register file with no
//   spills.  B fragments are shared between the two q-subtile waves through
//   L0/L2 (a_bf + pv_bf = 64 MB, resident in the 192 MB L2).
// ---------------------------------------------------------------------------
__global__ void __launch_bounds__(128, 1)
lag_score_kernel(const __bf16* __restrict__ A,
                 const __bf16* __restrict__ PV,
                 float* __restrict__ out) {
    const int lane = threadIdx.x & 31;
    const int wv   = threadIdx.x >> 5;
    const int half = lane >> 4;
    const int l16  = lane & 15;
    const int qb   = blockIdx.x * 32 + (wv & 1) * 16;   // wave's 16-row q band
    const int cb   = blockIdx.y * 32 + (wv >> 1) * 16;  // wave's 16-col c band
    const int q    = qb + l16;
    const int c    = cb + l16;

    v8f acc[NLAG] = {};

    for (int d0 = 0; d0 < DH; d0 += 32) {
        for (int t = 0; t < Tn; ++t) {
            // A fragment for this q-subtile
            const __bf16* ap = A + ((size_t)q * Tn + t) * DH + d0 + half * 16;
            BfPair p;
            p.lo = *(const v8bf*)(ap);
            p.hi = *(const v8bf*)(ap + 8);
            v16bf af = __builtin_bit_cast(v16bf, p);

            // 9 shifted B fragments
            #pragma unroll
            for (int dl = 0; dl < NLAG; ++dl) {
                const int tp = (t - (dl - 4) + Tn) & (Tn - 1);   // (t - delta) mod T
                v16bf bf = *(const v16bf*)(PV + ((size_t)c * Tn + tp) * DH
                                           + d0 + half * 16);
                acc[dl] = __builtin_amdgcn_wmma_f32_16x16x32_bf16(
                    false, af, false, bf, (short)0, acc[dl], false, false);
            }
        }
    }

    // elementwise max over lags, store fp32 [Q,C]
    v8f best = acc[0];
    #pragma unroll
    for (int dl = 1; dl < NLAG; ++dl)
        #pragma unroll
        for (int r = 0; r < 8; ++r)
            best[r] = fmaxf(best[r], acc[dl][r]);
    #pragma unroll
    for (int r = 0; r < 8; ++r) {
        const int qrow = qb + r + 8 * half;
        out[(size_t)qrow * Cn + c] = best[r];
    }
}

// ---------------------------------------------------------------------------
// Launch
// ---------------------------------------------------------------------------
extern "C" void kernel_launch(void* const* d_in, const int* in_sizes, int n_in,
                              void* d_out, int out_size, void* d_ws, size_t ws_size,
                              hipStream_t stream) {
    (void)in_sizes; (void)n_in; (void)out_size; (void)ws_size;

    const float* audio = (const float*)d_in[0];
    const float* video = (const float*)d_in[1];
    const float* a_w1  = (const float*)d_in[2];
    const float* a_b1  = (const float*)d_in[3];
    const float* a_w2  = (const float*)d_in[4];
    const float* a_b2  = (const float*)d_in[5];
    const float* v_w1  = (const float*)d_in[6];
    const float* v_b1  = (const float*)d_in[7];
    const float* v_w2  = (const float*)d_in[8];
    const float* v_b2  = (const float*)d_in[9];
    const float* Wmat  = (const float*)d_in[10];
    // d_in[11] = max_lag (hardcoded 4 -> NLAG=9)

    // workspace layout
    const size_t ACT_BYTES = (size_t)NROWS * DH * sizeof(__bf16);   // 32 MB
    char* ws = (char*)d_ws;
    __bf16* a_bf  = (__bf16*)(ws);                      // a activations (bf16)
    __bf16* pv_bf = (__bf16*)(ws + ACT_BYTES);          // projected video (bf16)
    __bf16* h_bf  = (__bf16*)(ws + 2 * ACT_BYTES);      // hidden scratch (bf16)
    float*  Wc    = (float*)(ws + 3 * ACT_BYTES);       // folded weight 256x256
    float*  bc    = (float*)(ws + 3 * ACT_BYTES + (size_t)DH * DH * sizeof(float));

    // 1) fold W into video second layer
    fuse_w_kernel<<<dim3(DH), dim3(DH), 0, stream>>>(v_w2, v_b2, Wmat, Wc, bc);

    const dim3 gemm_grid(NROWS / 64, DH / 64);   // 1024 x 4
    const dim3 gemm_blk(256);

    // 2) audio MLP: relu(audio@a_w1+b1) -> h ; h@a_w2+b2 -> a_bf
    gemm_bias_act<false, true ><<<gemm_grid, gemm_blk, 0, stream>>>(
        (const void*)audio, a_w1, a_b1, h_bf, NROWS, DIN);
    gemm_bias_act<true,  false><<<gemm_grid, gemm_blk, 0, stream>>>(
        (const void*)h_bf, a_w2, a_b2, a_bf, NROWS, DH);

    // 3) video MLP with folded projection: relu(video@v_w1+b1) -> h ; h@Wc+bc -> pv_bf
    gemm_bias_act<false, true ><<<gemm_grid, gemm_blk, 0, stream>>>(
        (const void*)video, v_w1, v_b1, h_bf, NROWS, DIN);
    gemm_bias_act<true,  false><<<gemm_grid, gemm_blk, 0, stream>>>(
        (const void*)h_bf, Wc, bc, pv_bf, NROWS, DH);

    // 4) lag-shifted correlation + max
    lag_score_kernel<<<dim3(Qn / 32, Cn / 32), dim3(128), 0, stream>>>(
        a_bf, pv_bf, (float*)d_out);
}
